// Policy_memory_58402965291132
// MI455X (gfx1250) — compile-verified, matrix operand order checked by reading, not run on hardware
//
#include <hip/hip_runtime.h>
#include <hip/hip_bf16.h>
#include <math.h>

#define T_STEPS 256
#define BATCH   2048
#define IN_DIM  300
#define HID     100
#define OUT_DIM 33
#define NPAD    336   // 3 * 112 (each gate block padded to 112, 16-aligned)
#define HIDP    112
#define KP1     320   // IN_DIM padded to mult of 32
#define KP2     128   // HID padded to mult of 32
#define RS      136   // LDS row stride in bf16 elems (128 + 8 pad -> bank spread)

typedef __attribute__((ext_vector_type(16))) __bf16 v16bf;
typedef __attribute__((ext_vector_type(8)))  __bf16 v8bf;
typedef __attribute__((ext_vector_type(8)))  float  v8f;

union V16 { v16bf v; v8bf h[2]; };

__device__ __forceinline__ float sigmoidf_(float x) { return 1.0f / (1.0f + __expf(-x)); }
// tanh via one fast exp; saturates correctly for large |x| (exp -> inf -> 1)
__device__ __forceinline__ float tanhf_(float x)    { return 1.0f - 2.0f / (1.0f + __expf(2.0f * x)); }

// ---------------------------------------------------------------------------
// Phase 0: pack weights into zero-padded bf16 matrices.
//   Win [NPAD][KP1]: rows n' = mat*112 + j (mat: 0=r,1=z,2=e), cols k<300 valid
//   Wrc [NPAD][KP2]: same row scheme, cols k<100 valid
// ---------------------------------------------------------------------------
__global__ __launch_bounds__(256) void prep_weights(
    const float* __restrict__ Wrx, const float* __restrict__ Wrh,
    const float* __restrict__ Wzx, const float* __restrict__ Wzh,
    const float* __restrict__ Wex, const float* __restrict__ Weh,
    __bf16* __restrict__ Win, __bf16* __restrict__ Wrc)
{
    int idx = blockIdx.x * 256 + threadIdx.x;
    const int nWin = NPAD * KP1;
    const int nWrc = NPAD * KP2;
    if (idx < nWin) {
        int n = idx / KP1, k = idx % KP1;
        int mat = n / HIDP, j = n % HIDP;
        float v = 0.0f;
        if (j < HID && k < IN_DIM) {
            const float* W = (mat == 0) ? Wrx : ((mat == 1) ? Wzx : Wex);
            v = W[j * IN_DIM + k];
        }
        Win[idx] = (__bf16)v;
    } else if (idx < nWin + nWrc) {
        int i2 = idx - nWin;
        int n = i2 / KP2, k = i2 % KP2;
        int mat = n / HIDP, j = n % HIDP;
        float v = 0.0f;
        if (j < HID && k < HID) {
            const float* W = (mat == 0) ? Wrh : ((mat == 1) ? Wzh : Weh);
            v = W[j * HID + k];
        }
        Wrc[i2] = (__bf16)v;
    }
}

// ---------------------------------------------------------------------------
// Phase 1: input projections  P[m][n'] = sum_k src[m][k] * Wcat[n'][k]
// WMMA orientation: A = weights (M-dim = n'), B = activations (N-dim = m).
// With the CDNA5 16-bit fragment layouts this makes every A-load, B-load and
// D-store a contiguous 16B access. 8 waves/block, each wave owns 16 rows.
// src rows preloaded once into 10 register fragments, reused over 21 n-tiles.
// ---------------------------------------------------------------------------
__global__ __launch_bounds__(256) void input_proj(
    const float*  __restrict__ src,   // [M][300] f32
    const __bf16* __restrict__ Win,   // [336][320] bf16
    __bf16*       __restrict__ P)     // [M][336] bf16
{
    const int wave = threadIdx.x >> 5;
    const int lane = threadIdx.x & 31;
    const int col  = lane & 15;   // activation row within tile / D column
    const int half = lane >> 4;   // which K half-run this lane owns
    const long m   = (long)blockIdx.x * 128 + wave * 16 + col;
    const float* row = src + m * IN_DIM;

    // Preload B fragments (activations, bf16) for all 10 K-steps.
    V16 bfrag[10];
    #pragma unroll
    for (int ks = 0; ks < 10; ++ks) {
        const int kb = ks * 32 + half * 8;
        float t[16];
        if (ks < 9) {
            const float4* p = (const float4*)(row + kb);
            float4 a0 = p[0], a1 = p[1], a2 = p[4], a3 = p[5];
            t[0]=a0.x;  t[1]=a0.y;  t[2]=a0.z;  t[3]=a0.w;
            t[4]=a1.x;  t[5]=a1.y;  t[6]=a1.z;  t[7]=a1.w;
            t[8]=a2.x;  t[9]=a2.y;  t[10]=a2.z; t[11]=a2.w;
            t[12]=a3.x; t[13]=a3.y; t[14]=a3.z; t[15]=a3.w;
        } else {
            // K-tail 288..319: predicated scalar loads, zero past k=300
            #pragma unroll
            for (int j = 0; j < 8; ++j) t[j]     = (kb + j      < IN_DIM) ? row[kb + j]      : 0.0f;
            #pragma unroll
            for (int j = 0; j < 8; ++j) t[8 + j] = (kb + 16 + j < IN_DIM) ? row[kb + 16 + j] : 0.0f;
        }
        #pragma unroll
        for (int j = 0; j < 16; ++j) bfrag[ks].v[j] = (__bf16)t[j];
    }

    #pragma unroll 3
    for (int nt = 0; nt < 21; ++nt) {
        v8f acc = {};
        #pragma unroll
        for (int ks = 0; ks < 10; ++ks) {
            const __bf16* w = Win + (nt * 16 + col) * KP1 + ks * 32 + half * 8;
            V16 a;
            a.h[0] = *(const v8bf*)w;
            a.h[1] = *(const v8bf*)(w + 16);
            acc = __builtin_amdgcn_wmma_f32_16x16x32_bf16(
                false, a.v, false, bfrag[ks].v, (short)0, acc, false, false);
        }
        // D: lane holds column m, rows n = nt*16 + half*8 + i -> contiguous store
        v8bf o;
        #pragma unroll
        for (int i = 0; i < 8; ++i) o[i] = (__bf16)acc[i];
        *(v8bf*)(P + m * NPAD + nt * 16 + half * 8) = o;
    }
}

// ---------------------------------------------------------------------------
// Phase 2: the serial recurrence. 4 waves/block, each wave owns 16 batch rows
// for all 257 steps (t==256 uses x's projections and emits next_hidden).
// h and h*r live in LDS (bf16, padded rows). Per step per wave: 3 gates *
// 7 n-tiles * 4 k-steps = 84 v_wmma_f32_16x16x32_bf16. The next step's P row
// is prefetched (global_prefetch_b8) while this step's WMMAs run.
// ---------------------------------------------------------------------------
__global__ __launch_bounds__(128) void recurrence(
    const __bf16* __restrict__ P,    // [B*T][336]
    const __bf16* __restrict__ Px,   // [B][336]
    const __bf16* __restrict__ Wrc,  // [336][128]
    float*        __restrict__ nh)   // [B][112] next_hidden (f32)
{
    __shared__ __bf16 hbuf[64 * RS];
    __shared__ __bf16 rbuf[64 * RS];
    for (int i = threadIdx.x; i < 64 * RS; i += 128) {
        hbuf[i] = (__bf16)0.0f;
        rbuf[i] = (__bf16)0.0f;
    }
    __syncthreads();

    const int wave = threadIdx.x >> 5;
    const int lane = threadIdx.x & 31;
    const int col  = lane & 15;
    const int half = lane >> 4;
    const int bloc = wave * 16 + col;               // local batch row
    const long b   = (long)blockIdx.x * 64 + bloc;  // global batch row
    __bf16* hrow = hbuf + bloc * RS;
    __bf16* rrow = rbuf + bloc * RS;

    float zv[7][8];

    for (int t = 0; t <= T_STEPS; ++t) {
        const __bf16* prow = (t < T_STEPS) ? (P + (b * T_STEPS + t) * NPAD)
                                           : (Px + b * NPAD);

        // Prefetch next step's projections while this step computes.
        {
            const __bf16* pn = (t + 1 < T_STEPS) ? (P + (b * T_STEPS + t + 1) * NPAD)
                                                 : (Px + b * NPAD);
            // two lanes share a row (half=0/1) -> cover both halves of the 672B row
            __builtin_prefetch(pn + half * 168, 0, 1);
            __builtin_prefetch(pn + 84 + half * 168, 0, 1);
        }

        // ---- pass 1: r gate, write h*r into rbuf ----
        #pragma unroll
        for (int nt = 0; nt < 7; ++nt) {
            v8f acc = {};
            #pragma unroll
            for (int ks = 0; ks < 4; ++ks) {
                const __bf16* w = Wrc + (nt * 16 + col) * KP2 + ks * 32 + half * 8;
                const __bf16* hp = hrow + ks * 32 + half * 8;
                V16 a, bb;
                a.h[0]  = *(const v8bf*)w;  a.h[1]  = *(const v8bf*)(w + 16);
                bb.h[0] = *(const v8bf*)hp; bb.h[1] = *(const v8bf*)(hp + 16);
                acc = __builtin_amdgcn_wmma_f32_16x16x32_bf16(
                    false, a.v, false, bb.v, (short)0, acc, false, false);
            }
            v8bf pv = *(const v8bf*)(prow + nt * 16 + half * 8);
            v8bf hv = *(const v8bf*)(hrow + nt * 16 + half * 8);
            v8bf hr;
            #pragma unroll
            for (int i = 0; i < 8; ++i) {
                float r = sigmoidf_(acc[i] + (float)pv[i]);
                hr[i] = (__bf16)((float)hv[i] * r);
            }
            *(v8bf*)(rrow + nt * 16 + half * 8) = hr;
        }

        // ---- pass 2: z gate (kept in registers) ----
        #pragma unroll
        for (int nt = 0; nt < 7; ++nt) {
            v8f acc = {};
            #pragma unroll
            for (int ks = 0; ks < 4; ++ks) {
                const __bf16* w = Wrc + (HIDP + nt * 16 + col) * KP2 + ks * 32 + half * 8;
                const __bf16* hp = hrow + ks * 32 + half * 8;
                V16 a, bb;
                a.h[0]  = *(const v8bf*)w;  a.h[1]  = *(const v8bf*)(w + 16);
                bb.h[0] = *(const v8bf*)hp; bb.h[1] = *(const v8bf*)(hp + 16);
                acc = __builtin_amdgcn_wmma_f32_16x16x32_bf16(
                    false, a.v, false, bb.v, (short)0, acc, false, false);
            }
            v8bf pv = *(const v8bf*)(prow + HIDP + nt * 16 + half * 8);
            #pragma unroll
            for (int i = 0; i < 8; ++i) zv[nt][i] = sigmoidf_(acc[i] + (float)pv[i]);
        }

        // ---- pass 3: candidate (uses h*r), then h update ----
        #pragma unroll
        for (int nt = 0; nt < 7; ++nt) {
            v8f acc = {};
            #pragma unroll
            for (int ks = 0; ks < 4; ++ks) {
                const __bf16* w = Wrc + (2 * HIDP + nt * 16 + col) * KP2 + ks * 32 + half * 8;
                const __bf16* rp = rrow + ks * 32 + half * 8;
                V16 a, bb;
                a.h[0]  = *(const v8bf*)w;  a.h[1]  = *(const v8bf*)(w + 16);
                bb.h[0] = *(const v8bf*)rp; bb.h[1] = *(const v8bf*)(rp + 16);
                acc = __builtin_amdgcn_wmma_f32_16x16x32_bf16(
                    false, a.v, false, bb.v, (short)0, acc, false, false);
            }
            v8bf pv = *(const v8bf*)(prow + 2 * HIDP + nt * 16 + half * 8);
            v8bf hv = *(const v8bf*)(hrow + nt * 16 + half * 8);
            float hn[8];
            #pragma unroll
            for (int i = 0; i < 8; ++i) {
                float hl = tanhf_(acc[i] + (float)pv[i]);
                float z  = zv[nt][i];
                // faithful to source: h_new = z*hl + (1-z) + h
                hn[i] = z * hl + (1.0f - z) + (float)hv[i];
            }
            if (t < T_STEPS) {
                v8bf o;
                #pragma unroll
                for (int i = 0; i < 8; ++i) o[i] = (__bf16)hn[i];
                *(v8bf*)(hrow + nt * 16 + half * 8) = o;
            } else {
                float* d = nh + b * HIDP + nt * 16 + half * 8;
                ((float4*)d)[0] = make_float4(hn[0], hn[1], hn[2], hn[3]);
                ((float4*)d)[1] = make_float4(hn[4], hn[5], hn[6], hn[7]);
            }
        }
    }
}

// ---------------------------------------------------------------------------
// Phase 3: decoder + softmax. Tiny (2048 x 33 x 400 MACs) -> plain VALU,
// float4-vectorized loads.
// ---------------------------------------------------------------------------
__global__ __launch_bounds__(64) void decoder(
    const float* __restrict__ x,    // [B][300]
    const float* __restrict__ nh,   // [B][112]
    const float* __restrict__ Wd,   // [33][400]
    const float* __restrict__ bd,   // [33]
    float*       __restrict__ out)  // [B][33]
{
    __shared__ float y[OUT_DIM];
    const int b = blockIdx.x;
    const int j = threadIdx.x;
    if (j < OUT_DIM) {
        const float4* wx = (const float4*)(Wd + j * (IN_DIM + HID));
        const float4* wh = (const float4*)(Wd + j * (IN_DIM + HID) + IN_DIM);
        const float4* xr = (const float4*)(x + (long)b * IN_DIM);
        const float4* hr = (const float4*)(nh + (long)b * HIDP);
        float s = bd[j];
        #pragma unroll 5
        for (int c = 0; c < IN_DIM / 4; ++c) {
            float4 a = xr[c], w = wx[c];
            s += a.x * w.x + a.y * w.y + a.z * w.z + a.w * w.w;
        }
        #pragma unroll 5
        for (int c = 0; c < HID / 4; ++c) {
            float4 a = hr[c], w = wh[c];
            s += a.x * w.x + a.y * w.y + a.z * w.z + a.w * w.w;
        }
        y[j] = fmaxf(s, 0.0f);
    }
    __syncthreads();
    if (j < OUT_DIM) {
        float mx = -1e30f;
        for (int c = 0; c < OUT_DIM; ++c) mx = fmaxf(mx, y[c]);
        float sum = 0.0f;
        for (int c = 0; c < OUT_DIM; ++c) sum += __expf(y[c] - mx);
        out[(long)b * OUT_DIM + j] = __expf(y[j] - mx) / sum;
    }
}

// ---------------------------------------------------------------------------
extern "C" void kernel_launch(void* const* d_in, const int* in_sizes, int n_in,
                              void* d_out, int out_size, void* d_ws, size_t ws_size,
                              hipStream_t stream)
{
    const float* states = (const float*)d_in[0];  // [2048][256][300]
    const float* x      = (const float*)d_in[1];  // [2048][300]
    const float* Wrx    = (const float*)d_in[2];
    const float* Wrh    = (const float*)d_in[3];
    const float* Wzx    = (const float*)d_in[4];
    const float* Wzh    = (const float*)d_in[5];
    const float* Wex    = (const float*)d_in[6];
    const float* Weh    = (const float*)d_in[7];
    const float* Wd     = (const float*)d_in[8];
    const float* bd     = (const float*)d_in[9];
    float* out = (float*)d_out;

    // workspace layout (all offsets multiples of 256 bytes)
    const size_t P_B  = (size_t)BATCH * T_STEPS * NPAD * 2;   // 352,321,536
    const size_t PX_B = (size_t)BATCH * NPAD * 2;             //   1,376,256
    const size_t WI_B = (size_t)NPAD * KP1 * 2;               //     215,040
    const size_t WR_B = (size_t)NPAD * KP2 * 2;               //      86,016
    const size_t NH_B = (size_t)BATCH * HIDP * 4;             //     917,504
    if (ws_size < P_B + PX_B + WI_B + WR_B + NH_B) return;

    char* ws = (char*)d_ws;
    __bf16* P   = (__bf16*)(ws);
    __bf16* Px  = (__bf16*)(ws + P_B);
    __bf16* Win = (__bf16*)(ws + P_B + PX_B);
    __bf16* Wrc = (__bf16*)(ws + P_B + PX_B + WI_B);
    float*  nh  = (float*) (ws + P_B + PX_B + WI_B + WR_B);

    // phase 0: pack padded bf16 weights
    const int prep_elems = NPAD * KP1 + NPAD * KP2;
    prep_weights<<<(prep_elems + 255) / 256, 256, 0, stream>>>(
        Wrx, Wrh, Wzx, Wzh, Wex, Weh, Win, Wrc);

    // phase 1: input projections for full history and for x
    input_proj<<<(BATCH * T_STEPS) / 128, 256, 0, stream>>>(states, Win, P);
    input_proj<<<BATCH / 128, 256, 0, stream>>>(x, Win, Px);

    // phase 2: 257-step recurrence, emits next_hidden
    recurrence<<<BATCH / 64, 128, 0, stream>>>(P, Px, Wrc, nh);

    // phase 3: decoder + softmax
    decoder<<<BATCH, 64, 0, stream>>>(x, nh, Wd, bd, out);
}
